// MultiHeadVideoAttention_58394375357336
// MI455X (gfx1250) — matrix operand order repaired and checked by reading, not compiled, not run on hardware
//
#include <hip/hip_runtime.h>

// ---------------------------------------------------------------------------
// MI455X / gfx1250 window-attention pipeline.
//  - GEMMs: v_wmma_f32_16x16x32_f16, tiles staged to LDS by the Tensor Data
//    Mover (tensor_load_to_lds + s_wait_tensorcnt), double buffered.
//  - Attention: per-(window,head) blocks, WMMA QK^T and PV, shfl softmax.
//   B=2, T=8, H=32, W=32 -> N=8192 tok/batch, C=1024, NH=16, HD=64, win=128
// ---------------------------------------------------------------------------

typedef __attribute__((ext_vector_type(16))) _Float16     v16h;
typedef __attribute__((ext_vector_type(8)))  _Float16     v8h;
typedef __attribute__((ext_vector_type(8)))  float        v8f;
typedef __attribute__((ext_vector_type(4)))  unsigned int v4u;
typedef __attribute__((ext_vector_type(8)))  int          v8i;
typedef __attribute__((ext_vector_type(4)))  int          v4i;

#define BATCH   2
#define NTOK    8192
#define CDIM    1024
#define C3      3072
#define NHEAD   16
#define HDIM    64
#define NWIN    64
#define WTOK    128
#define ATT_SCALE 0.125f

#define KB      32            // K-step = one WMMA K per staged tile
#define PITCH   40            // LDS row pitch in halves: 32 data + 8 pad (80B)
#define BM      64            // block tile M
#define BN      256           // block tile N

static __device__ __forceinline__ v16h join16(v8h lo, v8h hi) {
    v16h r;
#pragma unroll
    for (int i = 0; i < 8; ++i) { r[i] = lo[i]; r[i + 8] = hi[i]; }
    return r;
}
static __device__ __forceinline__ v16h ldv16(const _Float16* p) {
    return join16(*(const v8h*)p, *(const v8h*)(p + 8));
}

// ---------------------------------------------------------------------------
// TDM: stage a [rows x KB] f16 tile (row-major, leading dim `ld` elements)
// into LDS at byte offset lds_addr, padding each 64B row with 16B so the LDS
// row pitch is 80B (conflict-free ds_load_b128 fragment reads).
// D# layout per CDNA5 ISA ch.8 (group0: count/lds/global/type, group1: dims).
// Toolchain uses the 6-arg builtin: (v4u g0, v8i g1, v4i g2, v4i g3, v8i, cpol)
// ---------------------------------------------------------------------------
#if __has_builtin(__builtin_amdgcn_tensor_load_to_lds)
#define HAVE_TDM 1
static __device__ __forceinline__ void tdm_load_tile(
    unsigned lds_addr, const _Float16* gptr, unsigned rows, unsigned ld)
{
    const unsigned long long ga = (unsigned long long)(uintptr_t)gptr;
    v4u g0;
    g0[0] = 1u;                                   // count=1 (valid), user mode
    g0[1] = lds_addr;                             // LDS byte address
    g0[2] = (unsigned)ga;                         // global_addr[31:0]
    g0[3] = (unsigned)(ga >> 32) | (2u << 30);    // global_addr[56:32], type=2
    v8i g1;
    //        data_size=2B   pad_en     interval: 16dw   amount: 4dw (16B)
    g1[0] = (int)((1u << 16) | (1u << 20) | (3u << 22) | (3u << 25));
    g1[1] = (int)((ld & 0xFFFFu) << 16);                      // tensor_dim0 lo
    g1[2] = (int)((ld >> 16) | ((rows & 0xFFFFu) << 16));     // dim0 hi, dim1 lo
    g1[3] = (int)((rows >> 16) | ((unsigned)KB << 16));       // dim1 hi, tile_dim0
    g1[4] = (int)(rows & 0xFFFFu);                            // tile_dim1 (tile_dim2=0)
    g1[5] = (int)ld;                                          // tensor_dim0_stride lo
    g1[6] = 0;                                                // stride0 hi, stride1 lo
    g1[7] = 0;
    const v4i z4 = {0, 0, 0, 0};                  // groups 2/3 unused (2D tensor)
    const v8i z8 = {0, 0, 0, 0, 0, 0, 0, 0};
    __builtin_amdgcn_tensor_load_to_lds(g0, g1, z4, z4, z8, 0);
}
#else
#define HAVE_TDM 0
#endif

// ---------------------------------------------------------------------------
// fp32 -> fp16
// ---------------------------------------------------------------------------
__global__ void cvt_f32_f16(const float* __restrict__ in, _Float16* __restrict__ out, int n) {
    int i = blockIdx.x * 256 + threadIdx.x;
    if (i < n) out[i] = (_Float16)in[i];
}

// ---------------------------------------------------------------------------
// C[M,N] = A[M,K] @ Bm[N,K]^T + bias[N]
// Block = 8 waves = 64x256 tile; wave = 32x64 (2x4 WMMA accs).
// Per k-step: TDM stages A(64xKB) + B(256xKB) into padded LDS (double buffer),
// wave0 tracks TENSORcnt; all waves read fragments from LDS.
// ---------------------------------------------------------------------------
template <bool OUT_F16>
__global__ __launch_bounds__(256) void gemm_nt_wmma(
    const _Float16* __restrict__ A, const _Float16* __restrict__ Bm,
    const float* __restrict__ bias, void* __restrict__ Cout,
    int M, int N, int K)
{
    __shared__ _Float16 smem[2][(BM + BN) * PITCH];   // 2 x 25.6 KB

    const int lane = threadIdx.x & 31;
    const int wave = threadIdx.x >> 5;
    const int wm = wave >> 2, wn = wave & 3;
    const int nBlk = N >> 8;
    const int bm = blockIdx.x / nBlk, bn = blockIdx.x % nBlk;
    const int r15 = lane & 15, sel = lane >> 4;

    const _Float16* Ab = A  + (size_t)bm * BM * K;
    const _Float16* Bb = Bm + (size_t)bn * BN * K;

    // LDS byte offsets: low 32 bits of the generic address of a __shared__
    // object are its LDS offset (flat-aperture mapping).
    const unsigned ldsA[2] = {
        (unsigned)(uintptr_t)(void*)&smem[0][0],
        (unsigned)(uintptr_t)(void*)&smem[1][0] };
    const unsigned ldsB[2] = {
        (unsigned)(uintptr_t)(void*)&smem[0][BM * PITCH],
        (unsigned)(uintptr_t)(void*)&smem[1][BM * PITCH] };

    const bool issuer = (wave == 0);
    const int NS = K / KB;

#if HAVE_TDM
    if (issuer) {
        tdm_load_tile(ldsA[0], Ab, BM, (unsigned)K);
        tdm_load_tile(ldsB[0], Bb, BN, (unsigned)K);
    }
#endif

    v8f acc[2][4] = {};
    for (int s = 0; s < NS; ++s) {
        const int cur = s & 1;
#if HAVE_TDM
        if (issuer) {
            if (s + 1 < NS) {
                const _Float16* An = Ab + (size_t)(s + 1) * KB;
                const _Float16* Bn = Bb + (size_t)(s + 1) * KB;
                tdm_load_tile(ldsA[cur ^ 1], An, BM, (unsigned)K);
                tdm_load_tile(ldsB[cur ^ 1], Bn, BN, (unsigned)K);
                __builtin_amdgcn_s_wait_tensorcnt(2);  // current 2 tiles resident
            } else {
                __builtin_amdgcn_s_wait_tensorcnt(0);
            }
        }
#else
        // Fallback: cooperative copy (no padding math beyond PITCH)
        for (int i = threadIdx.x; i < (BM + BN) * KB / 8; i += 256) {
            const int row = i / (KB / 8), c8 = (i % (KB / 8)) * 8;
            const _Float16* src = (row < BM) ? (Ab + (size_t)row * K + s * KB + c8)
                                             : (Bb + (size_t)(row - BM) * K + s * KB + c8);
            *(v8h*)&smem[cur][row * PITCH + c8] = *(const v8h*)src;
        }
#endif
        __syncthreads();

        const _Float16* As = &smem[cur][0];
        const _Float16* Bs = &smem[cur][BM * PITCH];
        v16h af[2], bf[4];
#pragma unroll
        for (int t = 0; t < 2; ++t) {
            const _Float16* p = As + (wm * 32 + t * 16 + r15) * PITCH + sel * 8;
            af[t] = join16(*(const v8h*)p, *(const v8h*)(p + 16));
        }
#pragma unroll
        for (int u = 0; u < 4; ++u)
            bf[u] = join16(*(const v8h*)(Bs + (wn * 64 + u * 16 + r15) * PITCH + sel * 8), *(const v8h*)(Bs + (wn * 64 + u * 16 + r15) * PITCH + sel * 8 + 16));
#pragma unroll
        for (int t = 0; t < 2; ++t)
#pragma unroll
            for (int u = 0; u < 4; ++u)
                acc[t][u] = __builtin_amdgcn_wmma_f32_16x16x32_f16(
                    false, af[t], false, bf[u], (short)0, acc[t][u], false, false);
        __syncthreads();   // all reads done before buffer is overwritten
    }

    // C layout: VGPR r -> (row = r + sel*8, col = lane&15) per 16x16 tile
#pragma unroll
    for (int t = 0; t < 2; ++t)
#pragma unroll
        for (int u = 0; u < 4; ++u) {
            const int row = bm * BM + wm * 32 + t * 16 + sel * 8;
            const int col = bn * BN + wn * 64 + u * 16 + r15;
            const float bv = bias[col];
#pragma unroll
            for (int r = 0; r < 8; ++r) {
                const float v = acc[t][u][r] + bv;
                if (OUT_F16) ((_Float16*)Cout)[(size_t)(row + r) * N + col] = (_Float16)v;
                else         ((float*)   Cout)[(size_t)(row + r) * N + col] = v;
            }
        }
}

// ---------------------------------------------------------------------------
// Window attention: 1 block per (batch, window, head); 8 waves, each owns a
// 16-row Q tile of the 128x128 score matrix.
// ---------------------------------------------------------------------------
__global__ __launch_bounds__(256) void win_attn_wmma(
    const _Float16* __restrict__ qkv, _Float16* __restrict__ out)
{
    __shared__ _Float16 vt[HDIM][WTOK + 8];       // V^T : [dim][token]
    __shared__ _Float16 pbuf[8][16][WTOK + 8];    // per-wave softmax(P)

    const int lane = threadIdx.x & 31;
    const int wv   = threadIdx.x >> 5;
    int bid = blockIdx.x;
    const int head = bid & 15; bid >>= 4;
    const int win  = bid & 63; const int b = bid >> 6;
    const int wtb = win >> 4, whb = (win >> 2) & 3, wwb = win & 3;

    auto tok = [&](int m) -> int {
        const int it = m >> 6, ih = (m >> 3) & 7, iw = m & 7;
        return (wtb * 2 + it) * 1024 + (whb * 8 + ih) * 32 + (wwb * 8 + iw);
    };

    const size_t base = (size_t)b * NTOK * C3 + (size_t)head * HDIM;
    const int r15 = lane & 15, sel = lane >> 4;

    for (int i = threadIdx.x; i < WTOK * HDIM; i += 256) {
        const int tm = i >> 6, d = i & 63;
        vt[d][tm] = qkv[base + (size_t)tok(tm) * C3 + 2 * CDIM + d];
    }
    __syncthreads();

    const int mbase = wv * 16;
    const _Float16* qrow = qkv + base + (size_t)tok(mbase + r15) * C3;

    v8f sacc[8] = {};
#pragma unroll
    for (int ks = 0; ks < 2; ++ks) {
        const int k0 = sel * 8 + ks * 32;
        const v16h a = join16(*(const v8h*)(qrow + k0), *(const v8h*)(qrow + k0 + 16));
#pragma unroll
        for (int j = 0; j < 8; ++j) {
            const _Float16* krow = qkv + base + (size_t)tok(j * 16 + r15) * C3 + CDIM;
            const v16h bf = join16(*(const v8h*)(krow + k0), *(const v8h*)(krow + k0 + 16));
            sacc[j] = __builtin_amdgcn_wmma_f32_16x16x32_f16(
                false, a, false, bf, (short)0, sacc[j], false, false);
        }
    }

    float mx[8], sm[8];
#pragma unroll
    for (int r = 0; r < 8; ++r) {
        float m_ = -1e30f;
#pragma unroll
        for (int j = 0; j < 8; ++j) m_ = fmaxf(m_, sacc[j][r]);
#pragma unroll
        for (int off = 8; off >= 1; off >>= 1) m_ = fmaxf(m_, __shfl_xor(m_, off, 16));
        mx[r] = m_ * ATT_SCALE;
    }
#pragma unroll
    for (int r = 0; r < 8; ++r) {
        float s = 0.f;
#pragma unroll
        for (int j = 0; j < 8; ++j) {
            const float p = __expf(sacc[j][r] * ATT_SCALE - mx[r]);
            sacc[j][r] = p;
            s += p;
        }
#pragma unroll
        for (int off = 8; off >= 1; off >>= 1) s += __shfl_xor(s, off, 16);
        sm[r] = s;
    }

#pragma unroll
    for (int j = 0; j < 8; ++j)
#pragma unroll
        for (int r = 0; r < 8; ++r)
            pbuf[wv][r + sel * 8][j * 16 + r15] = (_Float16)sacc[j][r];

    v8f oacc[4] = {};
#pragma unroll
    for (int ks = 0; ks < 4; ++ks) {
        const int k0 = sel * 8 + ks * 32;
        const v16h a = join16(*(const v8h*)&pbuf[wv][r15][k0],
                              *(const v8h*)&pbuf[wv][r15][k0 + 16]);
#pragma unroll
        for (int u = 0; u < 4; ++u) {
            const v16h bf = join16(*(const v8h*)&vt[u * 16 + r15][k0], *(const v8h*)&vt[u * 16 + r15][k0 + 16]);
            oacc[u] = __builtin_amdgcn_wmma_f32_16x16x32_f16(
                false, a, false, bf, (short)0, oacc[u], false, false);
        }
    }

#pragma unroll
    for (int u = 0; u < 4; ++u)
#pragma unroll
        for (int r = 0; r < 8; ++r) {
            const int m = mbase + r + sel * 8;
            const int n = tok(m);
            const float v = oacc[u][r] / sm[r];
            out[((size_t)b * NTOK + n) * CDIM + head * HDIM + u * 16 + r15] = (_Float16)v;
        }
}

// ---------------------------------------------------------------------------
// Host orchestration
// ---------------------------------------------------------------------------
extern "C" void kernel_launch(void* const* d_in, const int* in_sizes, int n_in,
                              void* d_out, int out_size, void* d_ws, size_t ws_size,
                              hipStream_t stream) {
    const float* x      = (const float*)d_in[0];
    const float* qkv_w  = (const float*)d_in[1];
    const float* qkv_b  = (const float*)d_in[2];
    const float* proj_w = (const float*)d_in[3];
    const float* proj_b = (const float*)d_in[4];
    float* out = (float*)d_out;

    const size_t MROWS = (size_t)BATCH * NTOK;        // 16384
    _Float16* xh     = (_Float16*)d_ws;               // 16384*1024
    _Float16* qkvwh  = xh     + MROWS * CDIM;         // 3072*1024
    _Float16* projwh = qkvwh  + (size_t)C3 * CDIM;    // 1024*1024
    _Float16* qkvh   = projwh + (size_t)CDIM * CDIM;  // 16384*3072
    _Float16* attnh  = qkvh   + MROWS * C3;           // 16384*1024

    {
        int n = (int)(MROWS * CDIM);
        cvt_f32_f16<<<(n + 255) / 256, 256, 0, stream>>>(x, xh, n);
        n = C3 * CDIM;
        cvt_f32_f16<<<(n + 255) / 256, 256, 0, stream>>>(qkv_w, qkvwh, n);
        n = CDIM * CDIM;
        cvt_f32_f16<<<(n + 255) / 256, 256, 0, stream>>>(proj_w, projwh, n);
    }

    gemm_nt_wmma<true><<<dim3((MROWS / BM) * (C3 / BN)), 256, 0, stream>>>(
        xh, qkvwh, qkv_b, qkvh, (int)MROWS, C3, CDIM);

    win_attn_wmma<<<dim3(BATCH * NWIN * NHEAD), 256, 0, stream>>>(qkvh, attnh);

    gemm_nt_wmma<false><<<dim3((MROWS / BM) * (CDIM / BN)), 256, 0, stream>>>(
        attnh, projwh, proj_b, out, (int)MROWS, CDIM, CDIM);
}
